// VisualFusedEmbeddingLayer_21036749815974
// MI455X (gfx1250) — matrix operand order repaired and checked by reading, not compiled, not run on hardware
//
#include <hip/hip_runtime.h>
#include <float.h>
#include <math.h>

typedef __attribute__((ext_vector_type(16))) _Float16 v16h;
typedef __attribute__((ext_vector_type(8)))  float    v8f;

#define Bb   16
#define Nn   2048
#define Dd   512
#define Hh   512
#define Ee   1024
#define ROWS (Bb * Nn)      // 32768
#define KT16 16             // 512 / 32 k-tiles

union AF { v16h h16; float4 f4[2]; };

// ---------------------------------------------------------------- utilities
__device__ inline void atomicMaxFloat(float* addr, float val) {
  // order-independent float max via integer atomics (init must be -FLT_MAX)
  if (val >= 0.0f) atomicMax((int*)addr, __float_as_int(val));
  else             atomicMin((unsigned int*)addr, (unsigned int)__float_as_int(val));
}

// CDNA5 async copy global->LDS, 16B per lane, tracked by ASYNCcnt (ISA 10.x/15.18)
__device__ inline void async_copy_b128(unsigned lds_off, const _Float16* g) {
  asm volatile("global_load_async_to_lds_b128 %0, %1, off"
               :: "v"(lds_off), "v"(g) : "memory");
}
__device__ inline void wait_async0() {
  asm volatile("s_wait_asynccnt 0x0" ::: "memory");
}

// load 16-bit A fragment (16x32) for one wave; aptr = row base (+8 for hi lanes)
__device__ inline void load_a(AF& a, const _Float16* __restrict__ aptr, int kt) {
  a.f4[0] = *(const float4*)(aptr + kt * 32);
  a.f4[1] = *(const float4*)(aptr + kt * 32 + 16);
}
// B fragment from LDS panel: tile (j,kt) -> 512 halfs, 32B contiguous per lane
__device__ inline v16h load_b_lds(const _Float16* Bs, int j, int kt, int lane) {
  AF b;
  const _Float16* bp = Bs + (((j * KT16) + kt) << 9) + lane * 16;
  b.f4[0] = *(const float4*)(bp);
  b.f4[1] = *(const float4*)(bp + 8);
  return b.h16;
}

// ---------------------------------------------------------------- init
__global__ void init_pool(float* out2) {
  int i = blockIdx.x * 256 + threadIdx.x;
  if (i < Bb * Ee) out2[i] = -FLT_MAX;
}

// ------------------------------------------------- pack f32 weight -> WMMA-B f16 tiles
__global__ void pack_w(const float* __restrict__ W, _Float16* __restrict__ Wp, int Ncols) {
  int idx = blockIdx.x * 256 + threadIdx.x;
  if (idx >= Dd * Ncols) return;
  int k = idx / Ncols, n = idx - k * Ncols;
  int kt = k >> 5, kin = k & 31;
  int nt = n >> 4, nin = n & 15;
  int lane = (((kin >> 3) & 1) == 0) ? nin : (nin + 16);
  int e    = (kin & 7) + ((kin >> 4) << 3);
  Wp[(((size_t)(nt * KT16 + kt)) << 9) + lane * 16 + e] = (_Float16)W[idx];
}

// ------------------------------------------------- L2-normalize rows -> f16
__global__ __launch_bounds__(256) void l2norm_f16(const float* __restrict__ gf,
                                                  _Float16* __restrict__ xh) {
  __shared__ float red[256];
  const int t = threadIdx.x;
  const float* src = gf + (size_t)blockIdx.x * Dd;
  float a = src[t], b = src[t + 256];
  red[t] = a * a + b * b;
  __syncthreads();
  for (int off = 128; off > 0; off >>= 1) {
    if (t < off) red[t] += red[t + off];
    __syncthreads();
  }
  float inv = 1.0f / (sqrtf(red[0]) + 1e-8f);
  _Float16* dst = xh + (size_t)blockIdx.x * Dd;
  dst[t]       = (_Float16)(a * inv);
  dst[t + 256] = (_Float16)(b * inv);
}

// ------------------------------------------------- gf[:,0,:] -> f16 (raw, not normalized)
__global__ void gf_to_f16(const float* __restrict__ gf, _Float16* __restrict__ gfh) {
  int idx = blockIdx.x * 256 + threadIdx.x;   // 16*512
  if (idx >= Bb * Dd) return;
  int b = idx >> 9, c = idx & 511;
  gfh[idx] = (_Float16)gf[(size_t)b * Nn * Dd + c];
}

// ------------------------------------------------- shared GEMM body
// WG: 8 waves x (16 rows x 64 cols); B panel (64KB) staged to LDS via async DMA.
__device__ inline void gemm_tile(const _Float16* __restrict__ aptr,
                                 const _Float16* __restrict__ wpanel, // 64KB contiguous
                                 _Float16* Bs, v8f acc[4], int lane) {
  // stage B panel: 4096 chunks of 16B, 16 per thread, async to LDS
  unsigned ldsBase = (unsigned)(uintptr_t)Bs;
  {
    int t = threadIdx.x;
#pragma unroll
    for (int it = 0; it < 16; ++it) {
      int chunk = it * 256 + t;
      async_copy_b128(ldsBase + chunk * 16, wpanel + chunk * 8);
    }
  }
  // overlap: first A fragment from global while DMA is in flight
  AF a0, a1;
  load_a(a0, aptr, 0);
  wait_async0();
  __syncthreads();

  for (int kt = 0; kt < KT16; ++kt) {
    if (kt + 1 < KT16) {
      load_a(a1, aptr, kt + 1);
      if (kt + 2 < KT16) __builtin_prefetch(aptr + (kt + 2) * 32, 0, 3);
    }
#pragma unroll
    for (int j = 0; j < 4; ++j) {
      v16h bv = load_b_lds(Bs, j, kt, lane);
      acc[j] = __builtin_amdgcn_wmma_f32_16x16x32_f16(false, a0.h16, false, bv,
                                                      (short)0, acc[j], false, false);
    }
    a0 = a1;
  }
}

// ------------------------------------------------- GEMM1: h = x@W1 + b1  (f16 out)
__global__ __launch_bounds__(256) void gemm1_kernel(const _Float16* __restrict__ xh,
                                                    const _Float16* __restrict__ w1p,
                                                    const float* __restrict__ b1,
                                                    _Float16* __restrict__ h) {
  __shared__ _Float16 Bs[4 * KT16 * 512];           // 64KB
  const int wave = threadIdx.x >> 5, lane = threadIdx.x & 31;
  const int mBase = blockIdx.x * 128 + wave * 16;   // gridDim.x = 256
  const int ntB   = blockIdx.y * 4;                 // gridDim.y = 8 (64 cols)
  const int aoff  = (lane < 16) ? 0 : 8;
  const _Float16* aptr = xh + (size_t)(mBase + (lane & 15)) * Dd + aoff;

  v8f acc[4] = {};
  gemm_tile(aptr, w1p + ((size_t)ntB * KT16 << 9), Bs, acc, lane);

  const int rbase = mBase + ((lane < 16) ? 0 : 8);
#pragma unroll
  for (int j = 0; j < 4; ++j) {
    int col = (ntB + j) * 16 + (lane & 15);
    float bias = b1[col];
#pragma unroll
    for (int r = 0; r < 8; ++r)
      h[(size_t)(rbase + r) * Hh + col] = (_Float16)(acc[j][r] + bias);
  }
}

// ------------------------------------------------- BN stats: fixed-order two stage
__global__ __launch_bounds__(256) void stats_partial(const _Float16* __restrict__ h,
                                                     float* __restrict__ parts) {
  const int blk = blockIdx.x;          // 256 blocks, 128 rows each
  const int c0  = threadIdx.x * 2;
  const _Float16* p = h + (size_t)blk * 128 * Hh + c0;
  float s0 = 0.f, s1 = 0.f, q0 = 0.f, q1 = 0.f;
  for (int r = 0; r < 128; ++r) {
    float a = (float)p[0], b = (float)p[1];
    s0 += a; s1 += b; q0 += a * a; q1 += b * b;
    p += Hh;
  }
  float* pp = parts + blk * 1024;
  pp[c0] = s0; pp[c0 + 1] = s1;
  pp[512 + c0] = q0; pp[512 + c0 + 1] = q1;
}

__global__ __launch_bounds__(512) void stats_final(const float* __restrict__ parts,
                                                   const float* __restrict__ gamma,
                                                   const float* __restrict__ beta,
                                                   float* __restrict__ ss) {
  const int c = threadIdx.x;           // 512
  float s = 0.f, q = 0.f;
  for (int b = 0; b < 256; ++b) { s += parts[b * 1024 + c]; q += parts[b * 1024 + 512 + c]; }
  float mu  = s * (1.0f / (float)ROWS);
  float var = q * (1.0f / (float)ROWS) - mu * mu;
  float sc  = gamma[c] * rsqrtf(var + 1e-5f);
  ss[c]       = sc;
  ss[512 + c] = beta[c] - mu * sc;
}

// ------------------------------------------------- fused BN + ReLU (in place on h)
__global__ void bnrelu(_Float16* __restrict__ h, const float* __restrict__ ss) {
  size_t idx = (size_t)blockIdx.x * 256 + threadIdx.x;   // ROWS*Hh threads
  int c = (int)(idx & (Hh - 1));
  float v = (float)h[idx] * ss[c] + ss[512 + c];
  h[idx] = (_Float16)(v > 0.0f ? v : 0.0f);
}

// ------------------------------------------------- GEMM2 + max-pool over N
__global__ __launch_bounds__(256) void gemm2_kernel(const _Float16* __restrict__ hn,
                                                    const _Float16* __restrict__ w2p,
                                                    float* __restrict__ pool) {
  __shared__ _Float16 Bs[4 * KT16 * 512];            // 64KB
  const int wave = threadIdx.x >> 5, lane = threadIdx.x & 31;
  const int b    = blockIdx.z;                       // 16 batches
  const int mIn  = blockIdx.x * 128 + wave * 16;     // gridDim.x = 16 (rows in batch)
  const int ntB  = blockIdx.y * 4;                   // gridDim.y = 16 (64 cols)
  const int aoff = (lane < 16) ? 0 : 8;
  const _Float16* aptr = hn + (size_t)(b * Nn + mIn + (lane & 15)) * Hh + aoff;

  v8f acc[4] = {};
  gemm_tile(aptr, w2p + ((size_t)ntB * KT16 << 9), Bs, acc, lane);

#pragma unroll
  for (int j = 0; j < 4; ++j) {
    float m = acc[j][0];
#pragma unroll
    for (int r = 1; r < 8; ++r) m = fmaxf(m, acc[j][r]);
    m = fmaxf(m, __shfl_xor(m, 16, 32));             // combine row-halves (same column)
    if (lane < 16)
      atomicMaxFloat(&pool[b * Ee + (ntB + j) * 16 + lane], m);
  }
}

// ------------------------------------------------- fused = pool+b2 + gf@Wlin + blin
__global__ __launch_bounds__(256) void final_kernel(const _Float16* __restrict__ gfh,
                                                    const _Float16* __restrict__ wlp,
                                                    const float* __restrict__ blin,
                                                    const float* __restrict__ b2,
                                                    float* __restrict__ out) {
  const int wave = threadIdx.x >> 5, lane = threadIdx.x & 31;
  const int nt   = blockIdx.x * 8 + wave;            // 64 n-tiles total
  const int aoff = (lane < 16) ? 0 : 8;
  const _Float16* aptr = gfh + (size_t)(lane & 15) * Dd + aoff;   // M = 16 batches

  v8f acc = {};
  for (int kt = 0; kt < KT16; ++kt) {
    AF a;
    load_a(a, aptr, kt);
    AF bfr;
    const _Float16* bp = wlp + (((size_t)(nt * KT16 + kt)) << 9) + lane * 16;
    bfr.f4[0] = *(const float4*)(bp);
    bfr.f4[1] = *(const float4*)(bp + 8);
    acc = __builtin_amdgcn_wmma_f32_16x16x32_f16(false, a.h16, false, bfr.h16,
                                                 (short)0, acc, false, false);
  }
  const int col   = nt * 16 + (lane & 15);
  const int rbase = (lane < 16) ? 0 : 8;
  float* pool = out + Bb * Ee;
#pragma unroll
  for (int r = 0; r < 8; ++r) {
    int bi = rbase + r;                               // batch index
    float lpv = pool[bi * Ee + col] + b2[col];        // fold b2 into pooled max
    out[bi * Ee + col]  = acc[r] + blin[col] + lpv;   // fused
    pool[bi * Ee + col] = lpv;                        // finalized local_pooled
  }
}

// ---------------------------------------------------------------- launch
extern "C" void kernel_launch(void* const* d_in, const int* in_sizes, int n_in,
                              void* d_out, int out_size, void* d_ws, size_t ws_size,
                              hipStream_t stream) {
  const float* gfeatures = (const float*)d_in[0];
  // d_in[1] base_features, d_in[2] atten: dead in the reference -> never read
  const float* W1    = (const float*)d_in[3];
  const float* b1    = (const float*)d_in[4];
  const float* gamma = (const float*)d_in[5];
  const float* beta  = (const float*)d_in[6];
  const float* W2    = (const float*)d_in[7];
  const float* b2    = (const float*)d_in[8];
  const float* Wlin  = (const float*)d_in[9];
  const float* blin  = (const float*)d_in[10];
  float* out = (float*)d_out;

  char* ws = (char*)d_ws;
  size_t off = 0;
  _Float16* xh  = (_Float16*)(ws + off); off += (size_t)ROWS * Dd * 2;   // 32 MB
  _Float16* h   = (_Float16*)(ws + off); off += (size_t)ROWS * Hh * 2;   // 32 MB
  _Float16* w1p = (_Float16*)(ws + off); off += (size_t)Dd * Hh * 2;
  _Float16* w2p = (_Float16*)(ws + off); off += (size_t)Hh * Ee * 2;
  _Float16* wlp = (_Float16*)(ws + off); off += (size_t)Dd * Ee * 2;
  _Float16* gfh = (_Float16*)(ws + off); off += (size_t)Bb * Dd * 2;
  float*  parts = (float*)(ws + off);    off += 256 * 1024 * 4;
  float*  ss    = (float*)(ws + off);    off += 1024 * 4;

  init_pool<<<(Bb * Ee + 255) / 256, 256, 0, stream>>>(out + Bb * Ee);

  pack_w<<<(Dd * Hh + 255) / 256, 256, 0, stream>>>(W1,   w1p, Hh);
  pack_w<<<(Dd * Ee + 255) / 256, 256, 0, stream>>>(W2,   w2p, Ee);
  pack_w<<<(Dd * Ee + 255) / 256, 256, 0, stream>>>(Wlin, wlp, Ee);

  l2norm_f16<<<ROWS, 256, 0, stream>>>(gfeatures, xh);
  gf_to_f16<<<(Bb * Dd + 255) / 256, 256, 0, stream>>>(gfeatures, gfh);

  gemm1_kernel<<<dim3(ROWS / 128, Hh / 64, 1), 256, 0, stream>>>(xh, w1p, b1, h);

  stats_partial<<<256, 256, 0, stream>>>(h, parts);
  stats_final<<<1, 512, 0, stream>>>(parts, gamma, beta, ss);
  bnrelu<<<(unsigned)((size_t)ROWS * Hh / 256), 256, 0, stream>>>(h, ss);

  gemm2_kernel<<<dim3(Nn / 128, Ee / 64, Bb), 256, 0, stream>>>(h, w2p, out + Bb * Ee);

  final_kernel<<<8, 256, 0, stream>>>(gfh, wlp, blin, b2, out);
  (void)in_sizes; (void)n_in; (void)out_size; (void)ws_size; (void)off;
}
